// _Attention_16054587752974
// MI455X (gfx1250) — matrix operand order repaired and checked by reading, not compiled, not run on hardware
//
#include <hip/hip_runtime.h>
#include <stdint.h>

// Problem constants (B,H,W,DIM,NH) = (4,32,32,768,12), HD=64, HW=1024
#define ADIM 768
#define ANH  12
#define AHD  64
#define AHW  1024
#define ABAT 4
#define NHF  (ABAT * ANH)   // 48 flat heads

typedef __attribute__((ext_vector_type(16))) __bf16 v16bf;
typedef __attribute__((ext_vector_type(8)))  __bf16 v8bf;
typedef __attribute__((ext_vector_type(8)))  float  v8f;

__device__ __forceinline__ uint16_t f32_to_bf16(float f) {
  union { float f; uint32_t u; } v; v.f = f;
  uint32_t u = v.u;
  return (uint16_t)((u + 0x7FFFu + ((u >> 16) & 1u)) >> 16);   // RNE
}
__device__ __forceinline__ float bf16_to_f32(uint16_t h) {
  union { uint32_t u; float f; } v; v.u = ((uint32_t)h) << 16;
  return v.f;
}
__device__ __forceinline__ v8f vzero8() {
  v8f z = {0.f,0.f,0.f,0.f,0.f,0.f,0.f,0.f};
  return z;
}

// ---------------------------------------------------------------------------
// WMMA fragment loaders (wave32, 16x16x32 bf16).
//
// A operand, 16x32 (MxK), row-major source with leading dim `ld`:
//   lane L holds row m0+(L&15); elements 0..7  = K at k0 + (L>>4)*16 + 0..7
//                               elements 8..15 = K at k0 + (L>>4)*16 + 8 + 0..7
// (contiguous k order, shared with load_b_frag)
__device__ __forceinline__ v16bf load_a_frag(const uint16_t* base, int ld,
                                             int m0, int k0, int lane) {
  int row = m0 + (lane & 15);
  int kk  = k0 + ((lane >> 4) << 4);
  v8bf lo = *(const v8bf*)(base + row * ld + kk);
  v8bf hi = *(const v8bf*)(base + row * ld + kk + 8);
  v16bf r;
#pragma unroll
  for (int i = 0; i < 8; ++i) { r[i] = lo[i]; r[i + 8] = hi[i]; }
  return r;
}

// B operand, 32x16 (KxN).  Source holds B^T row-major: src[col][k], lead dim ld.
//   lane L holds column n0+(L&15); 16 contiguous K values at k0 + (L>>4)*16.
// (same contiguous k order as load_a_frag)
__device__ __forceinline__ v16bf load_b_frag(const uint16_t* base, int ld,
                                             int n0, int k0, int lane) {
  int col = n0 + (lane & 15);
  int kk  = k0 + ((lane >> 4) << 4);
  return *(const v16bf*)(base + col * ld + kk);
}

__device__ __forceinline__ v8f wmma_bf16(v16bf a, v16bf b, v8f c) {
  return __builtin_amdgcn_wmma_f32_16x16x32_bf16(false, a, false, b,
                                                 (short)0, c, false, false);
}

// ---------------------------------------------------------------------------
// K0: f32 -> bf16 conversion
__global__ void attn16054_cvt_bf16(const float* __restrict__ src,
                                   uint16_t* __restrict__ dst, int n) {
  int i = blockIdx.x * blockDim.x + threadIdx.x;
  if (i < n) dst[i] = f32_to_bf16(src[i]);
}

// ---------------------------------------------------------------------------
// K1: qkv = x @ qkv_w^T + qkv_b  (M=4096, N=2304, K=768), scatter to
//     Qb[hf][pos][hd], Kb[hf][pos][hd], Vt[hf][hd][pos]  (bf16)
__global__ __launch_bounds__(32)
void attn16054_qkv_gemm(const uint16_t* __restrict__ Xb,   // [4096][768]
                        const uint16_t* __restrict__ Wb,   // [2304][768]
                        const float*    __restrict__ bias, // [2304]
                        uint16_t* __restrict__ Qb,
                        uint16_t* __restrict__ Kb,
                        uint16_t* __restrict__ Vt) {
  int lane = threadIdx.x & 31;
  int m0 = blockIdx.x * 16;
  int n0 = blockIdx.y * 64;
  v8f acc[4] = {vzero8(), vzero8(), vzero8(), vzero8()};
  for (int k0 = 0; k0 < ADIM; k0 += 32) {
    v16bf a = load_a_frag(Xb, ADIM, m0, k0, lane);
#pragma unroll
    for (int t = 0; t < 4; ++t) {
      v16bf b = load_b_frag(Wb, ADIM, n0 + t * 16, k0, lane);
      acc[t] = wmma_bf16(a, b, acc[t]);
    }
  }
  int half = lane >> 4, nl = lane & 15;
#pragma unroll
  for (int t = 0; t < 4; ++t) {
    int n     = n0 + t * 16 + nl;
    int which = n / ADIM;            // 0=q,1=k,2=v
    int rem   = n - which * ADIM;
    int head  = rem >> 6;
    int hd    = rem & 63;
    float bv  = bias[n];
#pragma unroll
    for (int r = 0; r < 8; ++r) {
      int m   = m0 + r + half * 8;
      int bi  = m >> 10;
      int pos = m & 1023;
      int hf  = bi * ANH + head;
      uint16_t h = f32_to_bf16(acc[t][r] + bv);
      if (which == 0)      Qb[((size_t)hf * AHW + pos) * AHD + hd] = h;
      else if (which == 1) Kb[((size_t)hf * AHW + pos) * AHD + hd] = h;
      else                 Vt[((size_t)hf * AHD + hd) * AHW + pos] = h;
    }
  }
}

// ---------------------------------------------------------------------------
// K2: decomposed rel-pos terms.
//   relH[hf][qh][qw][kh] = sum_d q[hf][qh*32+qw][d] * rel_pos_h[qh-kh+31][d]
//   relW[hf][qh][qw][kw] = sum_d q[hf][qh*32+qw][d] * rel_pos_w[qw-kw+31][d]
__global__ void attn16054_relpos(const uint16_t* __restrict__ Qb,
                                 const float* __restrict__ rph,  // [63][64]
                                 const float* __restrict__ rpw,  // [63][64]
                                 float* __restrict__ relH,
                                 float* __restrict__ relW) {
  int idx = blockIdx.x * blockDim.x + threadIdx.x;
  if (idx >= NHF * 32 * 32 * 32) return;
  int k  = idx & 31;
  int qw = (idx >> 5) & 31;
  int qh = (idx >> 10) & 31;
  int hf = idx >> 15;
  const uint16_t* qrow = Qb + ((size_t)hf * AHW + qh * 32 + qw) * AHD;
  const float* rh = rph + (size_t)(qh - k + 31) * AHD;
  const float* rw = rpw + (size_t)(qw - k + 31) * AHD;
  float sh = 0.f, sw = 0.f;
#pragma unroll
  for (int d = 0; d < AHD; ++d) {
    float q = bf16_to_f32(qrow[d]);
    sh += q * rh[d];
    sw += q * rw[d];
  }
  relH[idx] = sh;
  relW[idx] = sw;
}

// ---------------------------------------------------------------------------
// K3: fused attention.  One wave per (flat head, 16-query tile).
// LDS: S f32[16][1024] | P bf16[16][1024] | bH f32[16][32] | bW f32[16][32] | rsum[16]
__global__ __launch_bounds__(32)
void attn16054_attn(const uint16_t* __restrict__ Qb,
                    const uint16_t* __restrict__ Kb,
                    const uint16_t* __restrict__ Vt,
                    const float* __restrict__ relH,
                    const float* __restrict__ relW,
                    uint16_t* __restrict__ Ob) {   // [4096][768]
  extern __shared__ char smem[];
  float*    S    = (float*)smem;                                  // 64 KB
  uint16_t* P    = (uint16_t*)(smem + 16 * 1024 * 4);             // 32 KB
  float*    bH   = (float*)(smem + 16 * 1024 * 4 + 16 * 1024 * 2);
  float*    bW   = bH + 16 * 32;
  float*    rsum = bW + 16 * 32;

  int lane = threadIdx.x & 31;
  int hf = blockIdx.x;          // 0..47
  int q0 = blockIdx.y * 16;     // query tile
  int half = lane >> 4, nl = lane & 15;
  const uint16_t* Qh = Qb + (size_t)hf * AHW * AHD;
  const uint16_t* Kh = Kb + (size_t)hf * AHW * AHD;
  const uint16_t* Vh = Vt + (size_t)hf * AHD * AHW;

  // stage this tile's bias rows (16 queries x (32 kh + 32 kw) f32)
  for (int i = lane; i < 16 * 32; i += 32) {
    int m = i >> 5, k = i & 31;
    int q = q0 + m;
    size_t base = (((size_t)hf * 32 + (q >> 5)) * 32 + (q & 31)) * 32 + k;
    bH[i] = relH[base];
    bW[i] = relW[base];
  }
  __syncthreads();

  // ---- Phase A: S = (Q K^T) * scale + bias -> LDS
  v16bf aq0 = load_a_frag(Qh, AHD, q0, 0, lane);
  v16bf aq1 = load_a_frag(Qh, AHD, q0, 32, lane);
  const float scale = 0.125f;   // 64^-0.5
  for (int nt = 0; nt < 64; ++nt) {
    v8f acc = vzero8();
    v16bf b0 = load_b_frag(Kh, AHD, nt * 16, 0, lane);
    v16bf b1 = load_b_frag(Kh, AHD, nt * 16, 32, lane);
    acc = wmma_bf16(aq0, b0, acc);
    acc = wmma_bf16(aq1, b1, acc);
    int k  = nt * 16 + nl;
    int kh = k >> 5, kw = k & 31;
#pragma unroll
    for (int r = 0; r < 8; ++r) {
      int m = r + half * 8;
      S[m * 1024 + k] = acc[r] * scale + bH[m * 32 + kh] + bW[m * 32 + kw];
    }
  }
  __syncthreads();

  // ---- Phase B: row softmax (row = nl, halves split columns; join via shfl 16)
  {
    int row = nl;
    const float* srow = S + row * 1024 + half * 512;
    float mx = -1e30f;
    for (int j = 0; j < 512; j += 4) {
      float4 s = *(const float4*)(srow + j);
      mx = fmaxf(mx, fmaxf(fmaxf(s.x, s.y), fmaxf(s.z, s.w)));
    }
    mx = fmaxf(mx, __shfl_xor(mx, 16, 32));
    float sum = 0.f;
    uint16_t* prow = P + row * 1024 + half * 512;
    for (int j = 0; j < 512; j += 4) {
      float4 s = *(const float4*)(srow + j);
      float e0 = __expf(s.x - mx), e1 = __expf(s.y - mx);
      float e2 = __expf(s.z - mx), e3 = __expf(s.w - mx);
      sum += (e0 + e1) + (e2 + e3);
      prow[j + 0] = f32_to_bf16(e0);
      prow[j + 1] = f32_to_bf16(e1);
      prow[j + 2] = f32_to_bf16(e2);
      prow[j + 3] = f32_to_bf16(e3);
    }
    sum += __shfl_xor(sum, 16, 32);
    if (half == 0) rsum[row] = sum;
  }
  __syncthreads();

  // ---- Phase C: O = P V  (A from LDS, B from transposed V)
  v8f o[4] = {vzero8(), vzero8(), vzero8(), vzero8()};
  for (int kb = 0; kb < 1024; kb += 32) {
    v16bf ap = load_a_frag(P, 1024, 0, kb, lane);
#pragma unroll
    for (int t = 0; t < 4; ++t) {
      v16bf bv = load_b_frag(Vh, AHW, t * 16, kb, lane);
      o[t] = wmma_bf16(ap, bv, o[t]);
    }
  }

  // ---- Phase D: normalize, store as bf16 into [b*1024+q][head*64+hd]
  int b = hf / ANH, head = hf % ANH;
#pragma unroll
  for (int t = 0; t < 4; ++t) {
    int c = head * 64 + t * 16 + nl;
#pragma unroll
    for (int r = 0; r < 8; ++r) {
      int m = r + half * 8;
      float val = o[t][r] / rsum[m];
      Ob[((size_t)b * AHW + q0 + m) * ADIM + c] = f32_to_bf16(val);
    }
  }
}

// ---------------------------------------------------------------------------
// K4: out = attnO @ proj_w^T + proj_b   (M=4096, N=768, K=768), f32 out
__global__ __launch_bounds__(32)
void attn16054_proj_gemm(const uint16_t* __restrict__ Ab,  // [4096][768]
                         const uint16_t* __restrict__ Wb,  // [768][768]
                         const float* __restrict__ bias,
                         float* __restrict__ out) {
  int lane = threadIdx.x & 31;
  int m0 = blockIdx.x * 16;
  int n0 = blockIdx.y * 64;
  v8f acc[4] = {vzero8(), vzero8(), vzero8(), vzero8()};
  for (int k0 = 0; k0 < ADIM; k0 += 32) {
    v16bf a = load_a_frag(Ab, ADIM, m0, k0, lane);
#pragma unroll
    for (int t = 0; t < 4; ++t) {
      v16bf b = load_b_frag(Wb, ADIM, n0 + t * 16, k0, lane);
      acc[t] = wmma_bf16(a, b, acc[t]);
    }
  }
  int half = lane >> 4, nl = lane & 15;
#pragma unroll
  for (int t = 0; t < 4; ++t) {
    int n = n0 + t * 16 + nl;
    float bv = bias[n];
#pragma unroll
    for (int r = 0; r < 8; ++r) {
      int m = m0 + r + half * 8;
      out[(size_t)m * ADIM + n] = acc[t][r] + bv;
    }
  }
}

// ---------------------------------------------------------------------------
extern "C" void kernel_launch(void* const* d_in, const int* in_sizes, int n_in,
                              void* d_out, int out_size, void* d_ws, size_t ws_size,
                              hipStream_t stream) {
  const float* x      = (const float*)d_in[0];
  const float* qkv_w  = (const float*)d_in[1];
  const float* qkv_b  = (const float*)d_in[2];
  const float* proj_w = (const float*)d_in[3];
  const float* proj_b = (const float*)d_in[4];
  const float* rph    = (const float*)d_in[5];
  const float* rpw    = (const float*)d_in[6];
  float* out = (float*)d_out;

  char* ws = (char*)d_ws;
  size_t off = 0;
  auto take = [&](size_t bytes) -> char* {
    char* p = ws + off;
    off += (bytes + 255) & ~(size_t)255;
    return p;
  };
  uint16_t* Xb   = (uint16_t*)take((size_t)4096 * 768 * 2);
  uint16_t* Wqb  = (uint16_t*)take((size_t)2304 * 768 * 2);
  uint16_t* Wpb  = (uint16_t*)take((size_t)768 * 768 * 2);
  uint16_t* Qb   = (uint16_t*)take((size_t)NHF * AHW * AHD * 2);
  uint16_t* Kb   = (uint16_t*)take((size_t)NHF * AHW * AHD * 2);
  uint16_t* Vt   = (uint16_t*)take((size_t)NHF * AHD * AHW * 2);
  float*    relH = (float*)take((size_t)NHF * 32 * 32 * 32 * 4);
  float*    relW = (float*)take((size_t)NHF * 32 * 32 * 32 * 4);
  uint16_t* Ob   = (uint16_t*)take((size_t)4096 * 768 * 2);

  // K0: bf16 conversions
  {
    int n0 = 4096 * 768, n1 = 2304 * 768, n2 = 768 * 768;
    attn16054_cvt_bf16<<<(n0 + 255) / 256, 256, 0, stream>>>(x, Xb, n0);
    attn16054_cvt_bf16<<<(n1 + 255) / 256, 256, 0, stream>>>(qkv_w, Wqb, n1);
    attn16054_cvt_bf16<<<(n2 + 255) / 256, 256, 0, stream>>>(proj_w, Wpb, n2);
  }
  // K1: QKV projection (WMMA)
  attn16054_qkv_gemm<<<dim3(4096 / 16, 2304 / 64), 32, 0, stream>>>(
      Xb, Wqb, qkv_b, Qb, Kb, Vt);
  // K2: rel-pos bias terms
  {
    int n = NHF * 32 * 32 * 32;
    attn16054_relpos<<<(n + 255) / 256, 256, 0, stream>>>(Qb, rph, rpw, relH, relW);
  }
  // K3: fused attention (WMMA + LDS softmax)
  {
    size_t smem = 16 * 1024 * 4 + 16 * 1024 * 2 + 2 * 16 * 32 * 4 + 16 * 4;
    attn16054_attn<<<dim3(NHF, AHW / 16), 32, smem, stream>>>(
        Qb, Kb, Vt, relH, relW, Ob);
  }
  // K4: output projection (WMMA)
  attn16054_proj_gemm<<<dim3(4096 / 16, 768 / 64), 32, 0, stream>>>(
      Ob, Wpb, proj_b, out);
}
